// BiDecoder_82403242541306
// MI455X (gfx1250) — matrix operand order repaired and checked by reading, not compiled, not run on hardware
//
#include <hip/hip_runtime.h>
#include <hip/hip_bf16.h>

typedef __attribute__((ext_vector_type(16))) _Float16 v16h;
typedef __attribute__((ext_vector_type(8)))  _Float16 v8h;
typedef __attribute__((ext_vector_type(8)))  float    v8f;

static constexpr int DIM     = 64;   // feature dim
static constexpr int RN      = 5;    // ratings
static constexpr int QSTRIDE = 72;   // halves per Ps row in LDS (64 padded -> bank-conflict free)
static constexpr int RSTRIDE = 36;   // dwords per (r,k) partial row (32 padded)
static constexpr int NWAVES  = 8;    // waves per block (256 threads)

__device__ __forceinline__ void cvt8(v16h& a, int off, float4 x, float4 y) {
    a[off + 0] = (_Float16)x.x; a[off + 1] = (_Float16)x.y;
    a[off + 2] = (_Float16)x.z; a[off + 3] = (_Float16)x.w;
    a[off + 4] = (_Float16)y.x; a[off + 5] = (_Float16)y.y;
    a[off + 6] = (_Float16)y.z; a[off + 7] = (_Float16)y.w;
}

__global__ __launch_bounds__(256) void bidecoder_wmma_kernel(
    const float* __restrict__ ufeat,   // (N_USERS, 64)
    const float* __restrict__ ifeat,   // (N_ITEMS, 64)
    const float* __restrict__ Ps,      // (5, 64, 64)
    const int*   __restrict__ src,     // (E,)
    const int*   __restrict__ dst,     // (E,)
    float*       __restrict__ out,     // (E,)
    int ntiles)                        // E/16 full tiles
{
    // Ps as f16, row-padded 64->72 halves: B-operand ds_load_b128 hits all 64 banks. 45 KB.
    __shared__ __align__(16) _Float16 Qs[RN * DIM * QSTRIDE];
    // per-wave stage-2 partials: [wave][r][k][32 lanes], row stride 36 dwords. 45 KB.
    __shared__ __align__(16) float Red[NWAVES * RN * 8 * RSTRIDE];

    const int tid = threadIdx.x;

    // ---- stage Ps -> f16 LDS (320 rows x 8 chunks of 8 floats) ----
    for (int t = tid; t < RN * DIM * 8; t += 256) {
        const int row = t >> 3;          // r*64 + d
        const int c   = t & 7;           // 8-float chunk
        const float4* sp = (const float4*)(Ps + row * DIM + c * 8);
        float4 f0 = sp[0], f1 = sp[1];
        v8h h;
        h[0] = (_Float16)f0.x; h[1] = (_Float16)f0.y;
        h[2] = (_Float16)f0.z; h[3] = (_Float16)f0.w;
        h[4] = (_Float16)f1.x; h[5] = (_Float16)f1.y;
        h[6] = (_Float16)f1.z; h[7] = (_Float16)f1.w;
        *(v8h*)&Qs[row * QSTRIDE + c * 8] = h;
    }
    __syncthreads();

    const int lane = tid & 31;
    const int p    = lane >> 4;    // lane half: K sub-chunk select in A/B layouts
    const int q16  = lane & 15;    // M row (A) / N column (B,C)
    const int wid  = tid >> 5;
    const int wave = blockIdx.x * NWAVES + wid;
    const int nwav = gridDim.x * NWAVES;
    float* redw = Red + wid * (RN * 8 * RSTRIDE);

    for (int tile = wave; tile < ntiles; tile += nwav) {
        const int base = tile * 16;

        // ---- A operands: item rows (gather by dst) in f16 A-layout ----
        // lane q16+16p holds row q16; halves 0..15 = K p*16..p*16+15
        const int     vrow = dst[base + q16];
        const float4* vp4  = (const float4*)(ifeat + (size_t)vrow * DIM);
        v16h a0, a1;
        cvt8(a0, 0, vp4[     p * 4], vp4[     p * 4 + 1]);
        cvt8(a0, 8, vp4[     p * 4 + 2], vp4[     p * 4 + 3]);
        cvt8(a1, 0, vp4[ 8 + p * 4], vp4[ 8 + p * 4 + 1]);
        cvt8(a1, 8, vp4[ 8 + p * 4 + 2], vp4[ 8 + p * 4 + 3]);

        // ---- U gathered directly in WMMA C-register layout ----
        // uc[n][k] = ufeat[src[base + k + 8p]][n*16 + q16]   (coalesced 64B row reads)
        const int4* sp4 = (const int4*)(src + base + p * 8);
        const int4 s0 = sp4[0], s1 = sp4[1];
        const int srows[8] = {s0.x, s0.y, s0.z, s0.w, s1.x, s1.y, s1.z, s1.w};
        float uc[4][8];
#pragma unroll
        for (int k = 0; k < 8; ++k) {
            const float* up = ufeat + (size_t)srows[k] * DIM + q16;
#pragma unroll
            for (int n = 0; n < 4; ++n) uc[n][k] = up[n * 16];
        }

        // ---- per-rating GEMM tile + fused stage-2 partial dot ----
        // r NOT unrolled: keeps one rating's B operands/C tiles live -> no spills.
#pragma unroll 1
        for (int r = 0; r < RN; ++r) {
            float sc[8];
#pragma unroll
            for (int k = 0; k < 8; ++k) sc[k] = 0.f;
#pragma unroll
            for (int n = 0; n < 4; ++n) {
                // B[K=f][N=q16] = Ps[r][d=n*16+N][f]; halves i = K p*16+i
                const int d = n * 16 + q16;
                const _Float16* bp = &Qs[(r * DIM + d) * QSTRIDE + p * 16];
                const v16h b0 = *(const v16h*)(bp);       // f =  0..31
                const v16h b1 = *(const v16h*)(bp + 32);  // f = 32..63
                v8f c = {0.f, 0.f, 0.f, 0.f, 0.f, 0.f, 0.f, 0.f};
                c = __builtin_amdgcn_wmma_f32_16x16x32_f16(
                        false, a0, false, b0, (short)0, c, false, false);
                c = __builtin_amdgcn_wmma_f32_16x16x32_f16(
                        false, a1, false, b1, (short)0, c, false, false);
#pragma unroll
                for (int k = 0; k < 8; ++k)
                    sc[k] = fmaf(uc[n][k], c[k], sc[k]);
            }
            // park partials in LDS (conflict-free: 32 lanes -> 32 consecutive dwords)
#pragma unroll
            for (int k = 0; k < 8; ++k)
                redw[(r * 8 + k) * RSTRIDE + lane] = sc[k];
        }

        // ---- transpose-reduce from LDS: lane e owns edge e (dup across halves) ----
        const int e  = lane & 15;
        const int kk = e & 7;
        const int pp = e >> 3;
        float s[RN];
#pragma unroll
        for (int r = 0; r < RN; ++r) {
            const float4* rp =
                (const float4*)&redw[(r * 8 + kk) * RSTRIDE + pp * 16];
            float4 x0 = rp[0], x1 = rp[1], x2 = rp[2], x3 = rp[3];
            float t0 = (x0.x + x0.y) + (x0.z + x0.w);
            float t1 = (x1.x + x1.y) + (x1.z + x1.w);
            float t2 = (x2.x + x2.y) + (x2.z + x2.w);
            float t3 = (x3.x + x3.y) + (x3.z + x3.w);
            s[r] = (t0 + t1) + (t2 + t3);
        }

        // ---- softmax over r + expected rating (per lane) ----
        float m = s[0];
#pragma unroll
        for (int r = 1; r < RN; ++r) m = fmaxf(m, s[r]);
        float se = 0.f, wv = 0.f;
#pragma unroll
        for (int r = 0; r < RN; ++r) {
            float ex = __expf(s[r] - m);
            se += ex;
            wv = fmaf(ex, (float)(r + 1), wv);
        }
        const float rat = wv * __frcp_rn(se);

        if (lane < 16) out[base + e] = rat;   // coalesced 64B store
    }
}

// Scalar fallback for E % 16 tail edges (E=1M divides by 16; safety only).
__global__ void bidecoder_tail_kernel(
    const float* __restrict__ ufeat, const float* __restrict__ ifeat,
    const float* __restrict__ Ps, const int* __restrict__ src,
    const int* __restrict__ dst, float* __restrict__ out,
    int start, int nedges)
{
    int e = start + blockIdx.x * blockDim.x + threadIdx.x;
    if (e >= nedges) return;
    const float* u = ufeat + (size_t)src[e] * DIM;
    const float* v = ifeat + (size_t)dst[e] * DIM;
    float s[RN];
    for (int r = 0; r < RN; ++r) {
        const float* P = Ps + (size_t)r * DIM * DIM;
        float acc = 0.f;
        for (int d = 0; d < DIM; ++d) {
            float t = 0.f;
            for (int f = 0; f < DIM; ++f) t = fmaf(P[d * DIM + f], v[f], t);
            acc = fmaf(u[d], t, acc);
        }
        s[r] = acc;
    }
    float m = s[0];
    for (int r = 1; r < RN; ++r) m = fmaxf(m, s[r]);
    float se = 0.f, wv = 0.f;
    for (int r = 0; r < RN; ++r) {
        float e2 = __expf(s[r] - m);
        se += e2;
        wv = fmaf(e2, (float)(r + 1), wv);
    }
    out[e] = wv / se;
}

extern "C" void kernel_launch(void* const* d_in, const int* in_sizes, int n_in,
                              void* d_out, int out_size, void* d_ws, size_t ws_size,
                              hipStream_t stream) {
    (void)n_in; (void)d_ws; (void)ws_size; (void)out_size;
    const float* ufeat = (const float*)d_in[0];
    const float* ifeat = (const float*)d_in[1];
    const float* Ps    = (const float*)d_in[2];
    const int*   src   = (const int*)d_in[3];
    const int*   dst   = (const int*)d_in[4];
    float* out = (float*)d_out;

    const int E      = in_sizes[3];
    const int ntiles = E / 16;
    const int rem    = E - ntiles * 16;

    if (ntiles > 0) {
        int blocks = (ntiles + NWAVES - 1) / NWAVES;
        if (blocks > 1024) blocks = 1024;   // grid-stride, ~7.6 tiles/wave at E=1M
        bidecoder_wmma_kernel<<<blocks, 256, 0, stream>>>(
            ufeat, ifeat, Ps, src, dst, out, ntiles);
    }
    if (rem > 0) {
        int tb = (rem + 63) / 64;
        bidecoder_tail_kernel<<<tb, 64, 0, stream>>>(
            ufeat, ifeat, Ps, src, dst, out, ntiles * 16, E);
    }
}
